// TransformerBlock_82772609729208
// MI455X (gfx1250) — compile-verified, matrix operand order checked by reading, not run on hardware
//
#include <hip/hip_runtime.h>

#ifndef __has_builtin
#define __has_builtin(x) 0
#endif

typedef __bf16 bf16;
typedef __attribute__((ext_vector_type(16))) __bf16 v16bf;
typedef __attribute__((ext_vector_type(8)))  __bf16 v8bf;
typedef __attribute__((ext_vector_type(8)))  float  v8f;

#define B_   2
#define T_   2048
#define D_   1024
#define H_   16
#define HD_  64
#define FF_  4096
#define MT_  (B_ * T_)   // 4096 rows total

union Frag16 { v16bf v; v8bf h[2]; };

// ---------------------------------------------------------------------------
// DPP16 row_xmask butterfly (xor within each 16-lane row) — replaces
// ds_bpermute-based __shfl_xor in the softmax reductions. All lanes valid,
// old = identity + bound_ctrl lets the DPP combiner fuse into the consumer.
// ---------------------------------------------------------------------------
template <int XM>
__device__ __forceinline__ float dpp_xor16(float v) {
  return __int_as_float(__builtin_amdgcn_update_dpp(
      0, __float_as_int(v), 0x160 + XM /*ROW_XMASK*/, 0xf, 0xf, true));
}
__device__ __forceinline__ float rowmax16(float v) {
  v = fmaxf(v, dpp_xor16<1>(v));
  v = fmaxf(v, dpp_xor16<2>(v));
  v = fmaxf(v, dpp_xor16<4>(v));
  v = fmaxf(v, dpp_xor16<8>(v));
  return v;
}
__device__ __forceinline__ float rowsum16(float v) {
  v += dpp_xor16<1>(v);
  v += dpp_xor16<2>(v);
  v += dpp_xor16<4>(v);
  v += dpp_xor16<8>(v);
  return v;
}

// ---------------------------------------------------------------------------
// CDNA5 async global->LDS fill (ASYNCcnt-tracked), via inline asm.
// vdst = VGPR holding LDS byte address; vaddr = VGPR pair with global address.
// ---------------------------------------------------------------------------
#define USE_ASYNC 1

__device__ __forceinline__ unsigned lds_offset(const void* p) {
  // LDS aperture: flat addr[31:0] == LDS byte offset (ISA 10.2)
  return (unsigned)(size_t)p;
}

__device__ __forceinline__ void async_b128(const void* lds_dst, const void* gsrc) {
  asm volatile("global_load_async_to_lds_b128 %0, %1, off"
               :
               : "v"(lds_offset(lds_dst)), "v"(gsrc)
               : "memory");
}

__device__ __forceinline__ void wait_async0() {
#if __has_builtin(__builtin_amdgcn_s_wait_asynccnt)
  __builtin_amdgcn_s_wait_asynccnt(0);
#else
  asm volatile("s_wait_asynccnt 0x0" ::: "memory");
#endif
}

// ---------------------------------------------------------------------------
// Weight transpose + fp32 -> bf16 cast:  in[R,C] fp32  ->  out[C,R] bf16
// ---------------------------------------------------------------------------
__global__ __launch_bounds__(256)
void castT_kernel(const float* __restrict__ in, bf16* __restrict__ out,
                  int R, int C) {
  size_t idx = (size_t)blockIdx.x * 256 + threadIdx.x;
  if (idx >= (size_t)R * C) return;
  int r = (int)(idx / C);
  int c = (int)(idx % C);
  out[(size_t)c * R + r] = (bf16)in[idx];
}

// ---------------------------------------------------------------------------
// LayerNorm (biased var, eps=1e-5), fp32 in -> bf16 out. One row per block.
// ---------------------------------------------------------------------------
__global__ __launch_bounds__(256)
void ln_kernel(const float* __restrict__ x, const float* __restrict__ alpha,
               const float* __restrict__ shift, bf16* __restrict__ out) {
  __shared__ float red[256];
  const int row = blockIdx.x;
  const int tid = threadIdx.x;
  const float* xr = x + (size_t)row * D_;

  float s = 0.f;
  for (int i = tid; i < D_; i += 256) s += xr[i];
  red[tid] = s; __syncthreads();
  for (int off = 128; off; off >>= 1) {
    if (tid < off) red[tid] += red[tid + off];
    __syncthreads();
  }
  const float mean = red[0] * (1.0f / D_);
  __syncthreads();

  float v = 0.f;
  for (int i = tid; i < D_; i += 256) { float d = xr[i] - mean; v += d * d; }
  red[tid] = v; __syncthreads();
  for (int off = 128; off; off >>= 1) {
    if (tid < off) red[tid] += red[tid + off];
    __syncthreads();
  }
  const float var  = red[0] * (1.0f / D_);
  const float rstd = rsqrtf(var + 1e-5f);

  for (int i = tid; i < D_; i += 256)
    out[(size_t)row * D_ + i] = (bf16)(alpha[i] * (xr[i] - mean) * rstd + shift[i]);
}

// ---------------------------------------------------------------------------
// bf16 WMMA GEMM:  C[M,N] = A[M,K] * Bt[N,K]^T   (Bt is K-major)
// Block tile 128x256, 8 waves (2 M x 4 N), wave tile 64x64.
// Double-buffered LDS, one barrier per K-step; async global->LDS fill.
// ---------------------------------------------------------------------------
enum { EPI_Q = 0, EPI_K = 1, EPI_V = 2, EPI_OPROJ = 3, EPI_FFN1 = 4, EPI_FFN2 = 5 };

template <int EPI>
__global__ __launch_bounds__(256)
void gemm_bf16_kernel(const bf16* __restrict__ A, const bf16* __restrict__ Bt,
                      int M, int N, int K,
                      const float* __restrict__ bias,
                      const float* __restrict__ resid,
                      float* __restrict__ outF, bf16* __restrict__ outH,
                      float scale) {
  __shared__ __attribute__((aligned(16))) bf16 lA[2][128 * 32];  // 16 KB
  __shared__ __attribute__((aligned(16))) bf16 lB[2][256 * 32];  // 32 KB

  const int tid  = threadIdx.x;
  const int lane = tid & 31;
  const int wave = tid >> 5;
  const int wm   = wave & 1;    // 0..1 : 64-row strip
  const int wn   = wave >> 1;   // 0..3 : 64-col strip
  const int bm   = blockIdx.y * 128;
  const int bn   = blockIdx.x * 256;
  const int half = lane >> 4;
  const int l16  = lane & 15;

  // per-thread staging coordinates (b128 chunks, K-contiguous)
  int ar[2], ac[2], br[4], bc[4];
#pragma unroll
  for (int j = 0; j < 2; ++j) { int c = tid + j * 256; ar[j] = c >> 2; ac[j] = (c & 3) * 8; }
#pragma unroll
  for (int j = 0; j < 4; ++j) { int c = tid + j * 256; br[j] = c >> 2; bc[j] = (c & 3) * 8; }

  const int nk = K >> 5;
  v8f acc[4][4] = {};

  auto compute = [&](int buf) {
    Frag16 af[4];
#pragma unroll
    for (int i = 0; i < 4; ++i) {
      int m = wm * 64 + i * 16 + l16;
      af[i].h[0] = *reinterpret_cast<const v8bf*>(&lA[buf][m * 32 + half * 8]);
      af[i].h[1] = *reinterpret_cast<const v8bf*>(&lA[buf][m * 32 + 16 + half * 8]);
    }
#pragma unroll
    for (int j = 0; j < 4; ++j) {
      int n = wn * 64 + j * 16 + l16;
      Frag16 bfg;
      bfg.h[0] = *reinterpret_cast<const v8bf*>(&lB[buf][n * 32 + half * 8]);
      bfg.h[1] = *reinterpret_cast<const v8bf*>(&lB[buf][n * 32 + 16 + half * 8]);
#pragma unroll
      for (int i = 0; i < 4; ++i)
        acc[i][j] = __builtin_amdgcn_wmma_f32_16x16x32_bf16(
            false, af[i].v, false, bfg.v, (short)0, acc[i][j], false, false);
    }
  };

#if USE_ASYNC
  // ---- async double-buffered pipeline (1 barrier / K-step) ----
  auto issue = [&](int kt, int buf) {
#pragma unroll
    for (int j = 0; j < 2; ++j)
      async_b128(&lA[buf][ar[j] * 32 + ac[j]],
                 &A[(size_t)(bm + ar[j]) * K + kt * 32 + ac[j]]);
#pragma unroll
    for (int j = 0; j < 4; ++j)
      async_b128(&lB[buf][br[j] * 32 + bc[j]],
                 &Bt[(size_t)(bn + br[j]) * K + kt * 32 + bc[j]]);
  };
  issue(0, 0);
  for (int kt = 0; kt < nk; ++kt) {
    wait_async0();          // tile kt landed (issued one iteration ago)
    __syncthreads();        // visible to all waves; prior reads of next buf done
    if (kt + 1 < nk) issue(kt + 1, (kt + 1) & 1);  // overlaps compute below
    compute(kt & 1);
  }
#else
  // ---- register-staged double-buffered pipeline (1 barrier / K-step) ----
  v8bf rA[2], rB[4];
  auto fetch = [&](int kt) {
#pragma unroll
    for (int j = 0; j < 2; ++j)
      rA[j] = *reinterpret_cast<const v8bf*>(&A[(size_t)(bm + ar[j]) * K + kt * 32 + ac[j]]);
#pragma unroll
    for (int j = 0; j < 4; ++j)
      rB[j] = *reinterpret_cast<const v8bf*>(&Bt[(size_t)(bn + br[j]) * K + kt * 32 + bc[j]]);
  };
  fetch(0);
  for (int kt = 0; kt < nk; ++kt) {
    const int buf = kt & 1;
#pragma unroll
    for (int j = 0; j < 2; ++j)
      *reinterpret_cast<v8bf*>(&lA[buf][ar[j] * 32 + ac[j]]) = rA[j];
#pragma unroll
    for (int j = 0; j < 4; ++j)
      *reinterpret_cast<v8bf*>(&lB[buf][br[j] * 32 + bc[j]]) = rB[j];
    __syncthreads();
    if (kt + 1 < nk) fetch(kt + 1);   // global loads overlap compute
    compute(buf);
  }
#endif

  // ---- fused epilogue --------------------------------------------------
#pragma unroll
  for (int i = 0; i < 4; ++i)
#pragma unroll
    for (int j = 0; j < 4; ++j)
#pragma unroll
      for (int r = 0; r < 8; ++r) {
        const int gm = bm + wm * 64 + i * 16 + r + half * 8;
        const int gn = bn + wn * 64 + j * 16 + l16;
        const float v = acc[i][j][r];
        if (EPI == EPI_Q || EPI == EPI_K) {
          int b = gm >> 11, t = gm & (T_ - 1);
          int h = gn >> 6,  hd = gn & (HD_ - 1);
          outH[((size_t)((b * H_ + h) * T_ + t)) * HD_ + hd] = (bf16)(v * scale);
        } else if (EPI == EPI_V) {
          int b = gm >> 11, t = gm & (T_ - 1);
          int h = gn >> 6,  hd = gn & (HD_ - 1);
          outH[((size_t)((b * H_ + h) * HD_ + hd)) * T_ + t] = (bf16)v;
        } else if (EPI == EPI_OPROJ) {
          outF[(size_t)gm * N + gn] = v + bias[gn] + resid[(size_t)gm * N + gn];
        } else if (EPI == EPI_FFN1) {
          float xg = v + bias[gn];
          float g  = 0.5f * xg *
                     (1.0f + tanhf(0.7978845608f * (xg + 0.044715f * xg * xg * xg)));
          outH[(size_t)gm * N + gn] = (bf16)g;
        } else { // EPI_FFN2
          outF[(size_t)gm * N + gn] = v + bias[gn] + resid[(size_t)gm * N + gn];
        }
      }
}

// ---------------------------------------------------------------------------
// Causal flash attention. One wave per (b,h, 32-query tile) = 2 q-subtiles
// sharing K/V fragments. Causal masking only on the diagonal key-block.
// q: bf16 [B,H,T,hd] (pre-scaled by 1/8), k: bf16 [B,H,T,hd],
// vT: bf16 [B,H,hd,T], ctx: bf16 [B,T,D]
// ---------------------------------------------------------------------------
__global__ __launch_bounds__(32)
void attn_kernel(const bf16* __restrict__ q, const bf16* __restrict__ k,
                 const bf16* __restrict__ vT, bf16* __restrict__ ctx) {
  __shared__ __attribute__((aligned(16))) bf16 lp[2][16 * 32];
  const int lane = threadIdx.x;
  const int l16  = lane & 15;
  const int half = lane >> 4;
  const int nqb  = T_ / 32;
  const int qb   = blockIdx.x % nqb;
  const int bh   = blockIdx.x / nqb;      // b*H + h
  const int b    = bh / H_;
  const int h    = bh % H_;
  const int q0   = qb * 32;               // multiple of 32

  // Q A-fragments: [q-subtile][hd chunk]
  Frag16 qA[2][2];
#pragma unroll
  for (int qt = 0; qt < 2; ++qt) {
    const bf16* qrow = q + ((size_t)bh * T_ + q0 + qt * 16 + l16) * HD_;
#pragma unroll
    for (int c = 0; c < 2; ++c) {
      qA[qt][c].h[0] = *reinterpret_cast<const v8bf*>(&qrow[c * 32 + half * 8]);
      qA[qt][c].h[1] = *reinterpret_cast<const v8bf*>(&qrow[c * 32 + 16 + half * 8]);
    }
  }

  v8f  O[2][4] = {};
  float mi[2][8], li[2][8];
#pragma unroll
  for (int qt = 0; qt < 2; ++qt)
#pragma unroll
    for (int r = 0; r < 8; ++r) { mi[qt][r] = -1e30f; li[qt][r] = 0.f; }

  auto process_block = [&](int kb, bool domask) {
    // ---- K fragments, shared by both q-subtiles ----
    Frag16 kF[2][2];   // [key subtile][hd chunk]
#pragma unroll
    for (int t = 0; t < 2; ++t) {
      const bf16* krow = k + ((size_t)bh * T_ + kb + t * 16 + l16) * HD_;
#pragma unroll
      for (int c = 0; c < 2; ++c) {
        kF[t][c].h[0] = *reinterpret_cast<const v8bf*>(&krow[c * 32 + half * 8]);
        kF[t][c].h[1] = *reinterpret_cast<const v8bf*>(&krow[c * 32 + 16 + half * 8]);
      }
    }
    // ---- scores ----
    v8f s[2][2];
#pragma unroll
    for (int qt = 0; qt < 2; ++qt)
#pragma unroll
      for (int t = 0; t < 2; ++t) {
        v8f z = {};
        z = __builtin_amdgcn_wmma_f32_16x16x32_bf16(false, qA[qt][0].v, false,
                                                    kF[t][0].v, (short)0, z, false, false);
        z = __builtin_amdgcn_wmma_f32_16x16x32_bf16(false, qA[qt][1].v, false,
                                                    kF[t][1].v, (short)0, z, false, false);
        s[qt][t] = z;
      }

    // ---- (diagonal only) causal mask + online softmax ----
#pragma unroll
    for (int qt = 0; qt < 2; ++qt) {
#pragma unroll
      for (int r = 0; r < 8; ++r) {
        if (domask) {
          const int qg = q0 + qt * 16 + r + half * 8;
#pragma unroll
          for (int t = 0; t < 2; ++t)
            if (kb + t * 16 + l16 > qg) s[qt][t][r] = -1e30f;
        }
        const float mv   = rowmax16(fmaxf(s[qt][0][r], s[qt][1][r]));
        const float mnew = fmaxf(mi[qt][r], mv);
        const float corr = __expf(mi[qt][r] - mnew);
        mi[qt][r] = mnew;
        const float p0 = __expf(s[qt][0][r] - mnew);
        const float p1 = __expf(s[qt][1][r] - mnew);
        const float rs = rowsum16(p0 + p1);
        li[qt][r] = li[qt][r] * corr + rs;
#pragma unroll
        for (int d = 0; d < 4; ++d) O[qt][d][r] *= corr;
        const int m = r + half * 8;
        lp[qt][m * 32 + l16]      = (bf16)p0;
        lp[qt][m * 32 + 16 + l16] = (bf16)p1;
      }
    }
    __syncthreads();   // single-wave WG: forces s_wait_dscnt, barrier ~NOP

    // ---- O += P V  (V fragments shared by both q-subtiles) ----
    Frag16 pA[2];
#pragma unroll
    for (int qt = 0; qt < 2; ++qt) {
      pA[qt].h[0] = *reinterpret_cast<const v8bf*>(&lp[qt][l16 * 32 + half * 8]);
      pA[qt].h[1] = *reinterpret_cast<const v8bf*>(&lp[qt][l16 * 32 + 16 + half * 8]);
    }
#pragma unroll
    for (int d = 0; d < 4; ++d) {
      const bf16* vrow = vT + ((size_t)bh * HD_ + d * 16 + l16) * T_;
      Frag16 vB;
      vB.h[0] = *reinterpret_cast<const v8bf*>(&vrow[kb + half * 8]);
      vB.h[1] = *reinterpret_cast<const v8bf*>(&vrow[kb + 16 + half * 8]);
#pragma unroll
      for (int qt = 0; qt < 2; ++qt)
        O[qt][d] = __builtin_amdgcn_wmma_f32_16x16x32_bf16(
            false, pA[qt].v, false, vB.v, (short)0, O[qt][d], false, false);
    }
    __syncthreads();
  };

  for (int kb = 0; kb < q0; kb += 32) process_block(kb, false);  // mask-free
  process_block(q0, true);                                       // diagonal

  // ---- normalize and emit ctx bf16 [B,T,D] ----
#pragma unroll
  for (int qt = 0; qt < 2; ++qt)
#pragma unroll
    for (int r = 0; r < 8; ++r) {
      const float inv = 1.0f / li[qt][r];
      const int t = q0 + qt * 16 + r + half * 8;
#pragma unroll
      for (int d = 0; d < 4; ++d)
        ctx[((size_t)(b * T_ + t)) * D_ + h * HD_ + d * 16 + l16] =
            (bf16)(O[qt][d][r] * inv);
    }
}

// ---------------------------------------------------------------------------
// Host-side orchestration
// ---------------------------------------------------------------------------
extern "C" void kernel_launch(void* const* d_in, const int* in_sizes, int n_in,
                              void* d_out, int out_size, void* d_ws, size_t ws_size,
                              hipStream_t stream) {
  const float* x  = (const float*)d_in[0];
  const float* wq = (const float*)d_in[1];
  const float* wk = (const float*)d_in[2];
  const float* wv = (const float*)d_in[3];
  const float* wo = (const float*)d_in[4];
  const float* bo = (const float*)d_in[5];
  const float* w1 = (const float*)d_in[6];
  const float* b1 = (const float*)d_in[7];
  const float* w2 = (const float*)d_in[8];
  const float* b2 = (const float*)d_in[9];
  const float* a1 = (const float*)d_in[10];
  const float* s1 = (const float*)d_in[11];
  const float* a2 = (const float*)d_in[12];
  const float* s2 = (const float*)d_in[13];

  char* ws = (char*)d_ws;
  const size_t MB = 1024ull * 1024ull;
  bf16* wqT = (bf16*)(ws + 0 * MB);     // [D,D]   2 MB
  bf16* wkT = (bf16*)(ws + 2 * MB);     // [D,D]   2 MB
  bf16* wvT = (bf16*)(ws + 4 * MB);     // [D,D]   2 MB
  bf16* woT = (bf16*)(ws + 6 * MB);     // [D,D]   2 MB
  bf16* w1T = (bf16*)(ws + 8 * MB);     // [FF,D]  8 MB
  bf16* w2T = (bf16*)(ws + 16 * MB);    // [D,FF]  8 MB
  bf16* h1  = (bf16*)(ws + 24 * MB);    // ln1(x)        8 MB
  bf16* qB  = (bf16*)(ws + 32 * MB);    // [B,H,T,hd]    8 MB
  bf16* kB  = (bf16*)(ws + 40 * MB);    // [B,H,T,hd]    8 MB
  bf16* vTB = (bf16*)(ws + 48 * MB);    // [B,H,hd,T]    8 MB
  bf16* ctx = (bf16*)(ws + 56 * MB);    // [B,T,D]       8 MB
  float* x1 = (float*)(ws + 64 * MB);   // residual fp32 16 MB
  bf16* h2  = (bf16*)(ws + 80 * MB);    // ln2(x1)       8 MB
  bf16* ffh = (bf16*)(ws + 88 * MB);    // gelu(ffn1)   32 MB

  // 1) weight cast+transpose (bf16, K-major)
  castT_kernel<<<(D_ * D_ + 255) / 256, 256, 0, stream>>>(wq, wqT, D_, D_);
  castT_kernel<<<(D_ * D_ + 255) / 256, 256, 0, stream>>>(wk, wkT, D_, D_);
  castT_kernel<<<(D_ * D_ + 255) / 256, 256, 0, stream>>>(wv, wvT, D_, D_);
  castT_kernel<<<(D_ * D_ + 255) / 256, 256, 0, stream>>>(wo, woT, D_, D_);
  castT_kernel<<<(D_ * FF_ + 255) / 256, 256, 0, stream>>>(w1, w1T, D_, FF_);
  castT_kernel<<<(FF_ * D_ + 255) / 256, 256, 0, stream>>>(w2, w2T, FF_, D_);

  // 2) pre-LN 1
  ln_kernel<<<MT_, 256, 0, stream>>>(x, a1, s1, h1);

  // 3) QKV projections (fused layout epilogues; Q pre-scaled by 1/sqrt(hd))
  dim3 gDD(D_ / 256, MT_ / 128);   // (4, 32)
  gemm_bf16_kernel<EPI_Q><<<gDD, 256, 0, stream>>>(h1, wqT, MT_, D_, D_,
      nullptr, nullptr, nullptr, qB, 0.125f);
  gemm_bf16_kernel<EPI_K><<<gDD, 256, 0, stream>>>(h1, wkT, MT_, D_, D_,
      nullptr, nullptr, nullptr, kB, 1.0f);
  gemm_bf16_kernel<EPI_V><<<gDD, 256, 0, stream>>>(h1, wvT, MT_, D_, D_,
      nullptr, nullptr, nullptr, vTB, 1.0f);

  // 4) causal flash attention (32 q-rows per wave)
  attn_kernel<<<B_ * H_ * (T_ / 32), 32, 0, stream>>>(qB, kB, vTB, ctx);

  // 5) output projection + bias + residual -> x1 (fp32)
  gemm_bf16_kernel<EPI_OPROJ><<<gDD, 256, 0, stream>>>(ctx, woT, MT_, D_, D_,
      bo, x, x1, nullptr, 1.0f);

  // 6) pre-LN 2
  ln_kernel<<<MT_, 256, 0, stream>>>(x1, a2, s2, h2);

  // 7) FFN1 + bias + GELU -> bf16
  dim3 gDF(FF_ / 256, MT_ / 128);  // (16, 32)
  gemm_bf16_kernel<EPI_FFN1><<<gDF, 256, 0, stream>>>(h2, w1T, MT_, FF_, D_,
      b1, nullptr, nullptr, ffh, 1.0f);

  // 8) FFN2 + bias + residual -> fp32 output
  gemm_bf16_kernel<EPI_FFN2><<<gDD, 256, 0, stream>>>(ffh, w2T, MT_, D_, FF_,
      b2, x1, (float*)d_out, nullptr, 1.0f);
}
